// TAGCNLayer_32899449488056
// MI455X (gfx1250) — matrix operand order repaired and checked by reading, not compiled, not run on hardware
//
#include <hip/hip_runtime.h>

#define N_NODES 10000   // == 625 * 16, so M tiles are exact
#define N_EDGES 640000
#define D_IN    128
#define D_OUT   128
#define D_FEAT  384     // (K_HOPS+1)*D_IN
#define LN_EPS  1e-5f

typedef __attribute__((ext_vector_type(16))) __bf16 v16bf;
typedef __attribute__((ext_vector_type(8)))  __bf16 v8bf;
typedef __attribute__((ext_vector_type(8)))  float  v8f;

// ---------------------------------------------------------------- utility

__global__ void zero_kernel(float* __restrict__ p, int n) {
    int i = blockIdx.x * blockDim.x + threadIdx.x;
    if (i < n) p[i] = 0.0f;
}

__global__ void deg_kernel(const int* __restrict__ dst, float* __restrict__ deg) {
    int e = blockIdx.x * blockDim.x + threadIdx.x;
    if (e < N_EDGES) atomicAdd(&deg[dst[e]], 1.0f);
}

__global__ void norm_kernel(float* __restrict__ deg) {
    int i = blockIdx.x * blockDim.x + threadIdx.x;
    if (i < N_NODES) deg[i] = rsqrtf(fmaxf(deg[i], 1.0f));
}

// One thread per (edge, 4-channel group): float4 gather from src row,
// scale by norm[src]*w, scatter-add into dst row. L2-resident working set.
__global__ void hop_kernel(const float* __restrict__ fin,
                           const float* __restrict__ norm,
                           const float* __restrict__ ew,
                           const int* __restrict__ src,
                           const int* __restrict__ dst,
                           float* __restrict__ fout) {
    long idx = (long)blockIdx.x * blockDim.x + threadIdx.x;  // e*32 + group
    if (idx >= (long)N_EDGES * 32) return;
    int e  = (int)(idx >> 5);
    int c4 = (int)(idx & 31) * 4;
    int s = src[e], d = dst[e];
    float scale = norm[s] * ew[e];
    const float4 v = *(const float4*)(fin + (long)s * D_IN + c4);
    float* o = fout + (long)d * D_IN + c4;
    atomicAdd(o + 0, v.x * scale);
    atomicAdd(o + 1, v.y * scale);
    atomicAdd(o + 2, v.z * scale);
    atomicAdd(o + 3, v.w * scale);
}

__global__ void scale_kernel(float* __restrict__ f, const float* __restrict__ norm) {
    int i = blockIdx.x * blockDim.x + threadIdx.x;
    if (i < N_NODES * D_IN) f[i] *= norm[i / D_IN];
}

// ------------------------------------------------------- bf16 pre-packing

// feats[m][k] = bf16 of [h | f1 | f2], 8 elements (one b128 store) per thread.
__global__ void pack_feats_kernel(const float* __restrict__ h,
                                  const float* __restrict__ f1,
                                  const float* __restrict__ f2,
                                  __bf16* __restrict__ feats) {
    int idx = blockIdx.x * blockDim.x + threadIdx.x;   // chunk-of-8 index
    if (idx >= N_NODES * (D_FEAT / 8)) return;
    int m  = idx / (D_FEAT / 8);
    int k8 = (idx % (D_FEAT / 8)) * 8;
    const float* s = (k8 < 128) ? (h  + (long)m * D_IN +  k8       )
                   : (k8 < 256) ? (f1 + (long)m * D_IN + (k8 - 128))
                                : (f2 + (long)m * D_IN + (k8 - 256));
    v8bf o;
#pragma unroll
    for (int j = 0; j < 8; ++j) o[j] = (__bf16)s[j];
    *(v8bf*)(feats + (long)m * D_FEAT + k8) = o;
}

__global__ void pack_w_kernel(const float* __restrict__ lw, __bf16* __restrict__ wb) {
    int idx = blockIdx.x * blockDim.x + threadIdx.x;   // chunk-of-8 index
    if (idx >= (D_OUT * D_FEAT) / 8) return;
    const float* s = lw + (long)idx * 8;
    v8bf o;
#pragma unroll
    for (int j = 0; j < 8; ++j) o[j] = (__bf16)s[j];
    *(v8bf*)(wb + (long)idx * 8) = o;
}

// ---------------------------------------------------------------- GEMM+LN

// Block = 256 threads (8 waves). blockIdx.x = M tile (16 rows, exact since
// 10000 % 16 == 0). Wave w owns N tile w. K=384 in 12 fully-unrolled WMMAs.
// A and B fragments per lane = two contiguous 16B bf16 runs each.
__global__ __launch_bounds__(256)
void gemm_ln_relu_kernel(const __bf16* __restrict__ feats,  // (10000, 384) bf16
                         const __bf16* __restrict__ wb,     // (128, 384)  bf16
                         const float* __restrict__ lb,
                         const float* __restrict__ gamma,
                         const float* __restrict__ beta,
                         float* __restrict__ out) {
    __shared__ float xs[16 * 128];

    const int tid   = threadIdx.x;
    const int wave  = tid >> 5;
    const int lane  = tid & 31;
    const int m0    = blockIdx.x * 16;
    const int n0    = wave * 16;
    const int lhalf = lane >> 4;       // 0 or 1
    const int l15   = lane & 15;

    const int m     = m0 + l15;        // A row for this lane (always valid)
    const int nlane = n0 + l15;        // B/C/D column for this lane

    const v8bf* arow = (const v8bf*)(feats + (long)m * D_FEAT);
    const v8bf* brow = (const v8bf*)(wb + (long)nlane * D_FEAT);
    __builtin_prefetch(brow, 0, 3);    // global_prefetch path

    v8f acc = {};
#pragma unroll
    for (int kt = 0; kt < 12; ++kt) {
        // A layout: elem i -> K = i + 8*((i>>3) + lhalf)
        //   i=0..7  : K = kt*32 + 8*lhalf      .. +7   (chunk kt*4 + lhalf)
        //   i=8..15 : K = kt*32 + 16 + 8*lhalf .. +7   (chunk kt*4 + 2 + lhalf)
        v8bf alo = arow[kt * 4 + lhalf];
        v8bf ahi = arow[kt * 4 + 2 + lhalf];
        // B layout: same as A, elem i -> K = i + 8*((i>>3) + lhalf)
        v8bf blo = brow[kt * 4 + lhalf];
        v8bf bhi = brow[kt * 4 + 2 + lhalf];
        v16bf a = __builtin_shufflevector(alo, ahi, 0,1,2,3,4,5,6,7,8,9,10,11,12,13,14,15);
        v16bf b = __builtin_shufflevector(blo, bhi, 0,1,2,3,4,5,6,7,8,9,10,11,12,13,14,15);
        acc = __builtin_amdgcn_wmma_f32_16x16x32_bf16(
            false, a, false, b, (short)0, acc, false, false);
    }

    // C/D layout: VGPR r holds row M = r + 8*lhalf, col N = lane&15
#pragma unroll
    for (int r = 0; r < 8; ++r) {
        int rr = r + 8 * lhalf;
        xs[rr * 128 + nlane] = acc[r] + lb[nlane];
    }
    __syncthreads();

    // Parallel LayerNorm: 16 threads per row, 8 channels each; xor-tree
    // reduction across the 16-lane group, then vectorized store with ReLU.
    const int row = tid >> 4;          // 0..15
    const int sub = tid & 15;          // 0..15
    const float* xr = xs + row * 128 + sub * 8;
    float s = 0.0f, sq = 0.0f;
#pragma unroll
    for (int j = 0; j < 8; ++j) { float v = xr[j]; s += v; sq += v * v; }
#pragma unroll
    for (int off = 1; off < 16; off <<= 1) {
        s  += __shfl_xor(s,  off, 16);
        sq += __shfl_xor(sq, off, 16);
    }
    float mu  = s * (1.0f / 128.0f);
    float var = sq * (1.0f / 128.0f) - mu * mu;
    float inv = rsqrtf(fmaxf(var, 0.0f) + LN_EPS);

    float ov[8];
#pragma unroll
    for (int j = 0; j < 8; ++j) {
        int c = sub * 8 + j;
        float v = (xr[j] - mu) * inv * gamma[c] + beta[c];
        ov[j] = v > 0.0f ? v : 0.0f;
    }
    float* orow = out + (long)(m0 + row) * D_OUT + sub * 8;
    *(float4*)(orow + 0) = make_float4(ov[0], ov[1], ov[2], ov[3]);
    *(float4*)(orow + 4) = make_float4(ov[4], ov[5], ov[6], ov[7]);
}

// ---------------------------------------------------------------- launch

extern "C" void kernel_launch(void* const* d_in, const int* in_sizes, int n_in,
                              void* d_out, int out_size, void* d_ws, size_t ws_size,
                              hipStream_t stream) {
    const float* h   = (const float*)d_in[0];
    const float* ew  = (const float*)d_in[1];
    const float* lw  = (const float*)d_in[2];
    const float* lb  = (const float*)d_in[3];
    const float* gam = (const float*)d_in[4];
    const float* bet = (const float*)d_in[5];
    const int*   src = (const int*)d_in[6];
    const int*   dst = (const int*)d_in[7];
    float*       out = (float*)d_out;

    char* ws = (char*)d_ws;
    float*  deg   = (float*)ws;                               // 40000 B (becomes norm)
    float*  f1    = (float*)(ws + 65536);                     // 5,120,000 B
    float*  f2    = (float*)(ws + 65536 + 5242880);           // 5,120,000 B
    __bf16* feats = (__bf16*)(ws + 65536 + 2 * 5242880);      // 7,680,000 B
    __bf16* wb    = (__bf16*)(ws + 65536 + 2 * 5242880 + 7680000); // 98,304 B

    const int NF = N_NODES * D_IN;

    zero_kernel<<<(N_NODES + 255) / 256, 256, 0, stream>>>(deg, N_NODES);
    zero_kernel<<<(NF + 255) / 256, 256, 0, stream>>>(f1, NF);
    zero_kernel<<<(NF + 255) / 256, 256, 0, stream>>>(f2, NF);

    deg_kernel<<<(N_EDGES + 255) / 256, 256, 0, stream>>>(dst, deg);
    norm_kernel<<<(N_NODES + 255) / 256, 256, 0, stream>>>(deg);

    const long hop_threads = (long)N_EDGES * 32;
    const unsigned hop_blocks = (unsigned)((hop_threads + 255) / 256);

    hop_kernel<<<hop_blocks, 256, 0, stream>>>(h, deg, ew, src, dst, f1);
    scale_kernel<<<(NF + 255) / 256, 256, 0, stream>>>(f1, deg);

    hop_kernel<<<hop_blocks, 256, 0, stream>>>(f1, deg, ew, src, dst, f2);
    scale_kernel<<<(NF + 255) / 256, 256, 0, stream>>>(f2, deg);

    pack_w_kernel<<<((D_OUT * D_FEAT / 8) + 255) / 256, 256, 0, stream>>>(lw, wb);
    pack_feats_kernel<<<((N_NODES * (D_FEAT / 8)) + 255) / 256, 256, 0, stream>>>(
        h, f1, f2, feats);

    gemm_ln_relu_kernel<<<N_NODES / 16, 256, 0, stream>>>(
        feats, wb, lb, gam, bet, out);
}
